// WeightOnlyLinear_61460982005824
// MI455X (gfx1250) — matrix operand order repaired and checked by reading, not compiled, not run on hardware
//
#include <hip/hip_runtime.h>

// ---------------------------------------------------------------------------
// Int4 weight-only linear:  out = x @ dequant(W)^T + bias
//   GEMM M=4096 (2*2048), N=4096, K=4096; W int4, group=128, f32 scale/zp.
// MI455X: v_wmma_f32_16x16x32_f16 with fp32 accumulators.
//   - 256x128 block tile, 8 waves, 64x64/wave (4x4 accs): 16 WMMA per
//     16 ds_load_b128 chains per K-step.
//   - f16 magic-bit dequant: as_f16(0x6400|nib) == 1024+nib exactly, so
//     (nib-zp)*s == (as_f16(0x6400|nib) + as_f16(0xE400|zp)) * (f16)s with
//     an exact packed add -> v_pk_add_f16/v_pk_mul_f16, ~35% less VALU that
//     would otherwise contend with the co-executing matrix pipe.
//   - Ping-pong LDS (one barrier per K-step) + register prefetch of the
//     next tile's globals overlapping the WMMA phase.
// ---------------------------------------------------------------------------

typedef __attribute__((ext_vector_type(16))) _Float16 v16h;
typedef __attribute__((ext_vector_type(2)))  _Float16 v2h;
typedef __attribute__((ext_vector_type(8)))  float    v8f;

#define K_DIM     4096
#define O_DIM     4096
#define GROUP     128
#define NGROUP    (K_DIM / GROUP)          // 32
#define ZPW       ((NGROUP + 7) / 8)       // 4
#define BM        256
#define BN        128
#define BK        32
#define NIT       (K_DIM / BK)             // 128
#define LDS_STR   40                       // 80B = 5*16B: odd*16B stride -> conflict-free, b128-aligned
#define A_ELEMS   (BM * LDS_STR)           // per buffer
#define B_ELEMS   (BN * LDS_STR)

union U32H2 { unsigned u; v2h h; };

__global__ __launch_bounds__(256)
void wqlinear_wmma_f16(const float* __restrict__ x,
                       const int*   __restrict__ pw,
                       const float* __restrict__ scale,
                       const int*   __restrict__ pzp,
                       const float* __restrict__ bias,
                       float* __restrict__ out)
{
    __shared__ _Float16 sA[2 * A_ELEMS];    // 2 x 20480 B
    __shared__ _Float16 sB[2 * B_ELEMS];    // 2 x 10240 B

    const int tid    = threadIdx.x;
    const int lane   = tid & 31;            // wave32
    const int wave   = tid >> 5;            // 0..7
    const int wave_m = wave >> 1;           // 0..3 -> 64 M rows
    const int wave_n = wave & 1;            // 0..1 -> 64 N cols
    const int half   = lane >> 4;
    const int l16    = lane & 15;

    const int block_n = blockIdx.x * BN;
    const int block_m = blockIdx.y * BM;

    // Loader roles: A -> thread t owns x-row t (32 floats/step);
    //               B -> thread t owns W-row t/2, half t&1 (16 nibbles/step).
    const int brow  = tid >> 1;
    const int bhalf = tid & 1;
    const float* xrow  = x + (size_t)(block_m + tid) * K_DIM;
    const int    wn    = block_n + brow;
    const int*   pwrow = pw + (size_t)wn * (K_DIM / 8) + bhalf * 2;

    const v8f vzero = {0.f,0.f,0.f,0.f,0.f,0.f,0.f,0.f};
    v8f acc[4][4];
#pragma unroll
    for (int i = 0; i < 4; ++i)
#pragma unroll
        for (int j = 0; j < 4; ++j) acc[i][j] = vzero;

    // ---- register prefetch state
    float4 pa[8];
    int2   pb;
    float  ps;
    int    pzv;

    auto load_regs = [&](int k) {
        const float4* src = (const float4*)(xrow + k);
#pragma unroll
        for (int v = 0; v < 8; ++v) pa[v] = src[v];
        const int g = k / GROUP;
        ps  = scale[(size_t)wn * NGROUP + g];
        pzv = (pzp[(size_t)wn * ZPW + (g >> 3)] >> ((g & 7) * 4)) & 15;
        pb  = *(const int2*)(pwrow + (k >> 3));
    };

    auto store_tile = [&](int buf) {
        // A: 32 floats -> 32 f16 (v_cvt_pk_f16_f32), 4x ds_store_b128
        {
            _Float16* dst = &sA[buf * A_ELEMS + tid * LDS_STR];
#pragma unroll
            for (int v = 0; v < 8; ++v) {
                dst[v * 4 + 0] = (_Float16)pa[v].x;
                dst[v * 4 + 1] = (_Float16)pa[v].y;
                dst[v * 4 + 2] = (_Float16)pa[v].z;
                dst[v * 4 + 3] = (_Float16)pa[v].w;
            }
        }
        // B: f16 magic-bit dequant, packed math, 2x ds_store_b128
        {
            const unsigned w0 = (unsigned)pb.x, w1 = (unsigned)pb.y;
            U32H2 z; z.u = 0xE400E400u + (unsigned)pzv * 0x00010001u; // -(1024+zp) x2
            const _Float16 sh = (_Float16)ps;
            const v2h sv = {sh, sh};
            uint4 q0, q1;
            unsigned* q0u = (unsigned*)&q0;
            unsigned* q1u = (unsigned*)&q1;
#pragma unroll
            for (int p = 0; p < 4; ++p) {
                const unsigned t0 = w0 >> (8 * p);
                const unsigned t1 = w1 >> (8 * p);
                U32H2 m0; m0.u = 0x64006400u | (t0 & 0xFu) | ((t0 << 12) & 0x000F0000u);
                U32H2 m1; m1.u = 0x64006400u | (t1 & 0xFu) | ((t1 << 12) & 0x000F0000u);
                U32H2 r0; r0.h = (m0.h + z.h) * sv;   // exact add, 1-rounding mul
                U32H2 r1; r1.h = (m1.h + z.h) * sv;
                q0u[p] = r0.u;
                q1u[p] = r1.u;
            }
            _Float16* dst = &sB[buf * B_ELEMS + brow * LDS_STR + bhalf * 16];
            *(uint4*)&dst[0] = q0;
            *(uint4*)&dst[8] = q1;
        }
    };

    // ---- pipeline prologue: tile 0 -> LDS buf 0, tile 1 -> regs
    load_regs(0);
    store_tile(0);
    load_regs(BK);
    __syncthreads();

    for (int it = 0; it < NIT; ++it) {
        const int cur = it & 1;
        const _Float16* cA = &sA[cur * A_ELEMS];
        const _Float16* cB = &sB[cur * B_ELEMS];

        // ---- A fragments (16x32 f16): lane = row l16;
        //      16B chunks at K elems 8*half and 16+8*half (ISA A layout)
        v16h afrag[4];
#pragma unroll
        for (int i = 0; i < 4; ++i) {
            const int row = wave_m * 64 + i * 16 + l16;
            union { v16h v; uint4 q[2]; } u;
            u.q[0] = *(const uint4*)&cA[row * LDS_STR + 8 * half];
            u.q[1] = *(const uint4*)&cA[row * LDS_STR + 16 + 8 * half];
            afrag[i] = u.v;
        }
        // ---- B fragments (32x16 f16): lane = col l16; 16B chunks at K elems 8*half and 16+8*half
#pragma unroll
        for (int j = 0; j < 4; ++j) {
            const int nrow = wave_n * 64 + j * 16 + l16;
            union { v16h v; uint4 q[2]; } u;
            u.q[0] = *(const uint4*)&cB[nrow * LDS_STR + 8 * half];
            u.q[1] = *(const uint4*)&cB[nrow * LDS_STR + 16 + 8 * half];
#pragma unroll
            for (int i = 0; i < 4; ++i) {
                acc[i][j] = __builtin_amdgcn_wmma_f32_16x16x32_f16(
                    false, afrag[i], false, u.v, (short)0, acc[i][j],
                    false, false);
            }
        }

        // ---- stage tile it+1 into the other buffer; prefetch tile it+2
        if (it + 1 < NIT) {
            store_tile(cur ^ 1);
            if (it + 2 < NIT) load_regs((it + 2) * BK);
        }
        __syncthreads();
    }

    // ---- epilogue: C layout -> VGPR r holds (m = r + 8*half, n = l16)
#pragma unroll
    for (int j = 0; j < 4; ++j) {
        const int   col = block_n + wave_n * 64 + j * 16 + l16;
        const float b   = bias[col];
#pragma unroll
        for (int i = 0; i < 4; ++i) {
            const int rowbase = block_m + wave_m * 64 + i * 16 + half * 8;
            float* op = out + (size_t)rowbase * O_DIM + col;
#pragma unroll
            for (int r = 0; r < 8; ++r) {
                op[(size_t)r * O_DIM] = acc[i][j][r] + b;
            }
        }
    }
}

extern "C" void kernel_launch(void* const* d_in, const int* in_sizes, int n_in,
                              void* d_out, int out_size, void* d_ws, size_t ws_size,
                              hipStream_t stream) {
    const float* x     = (const float*)d_in[0];
    const int*   pw    = (const int*)d_in[1];
    const float* scale = (const float*)d_in[2];
    const int*   pzp   = (const int*)d_in[3];
    const float* bias  = (const float*)d_in[4];
    float*       out   = (float*)d_out;

    const int M = in_sizes[0] / K_DIM;      // 4096
    dim3 grid(O_DIM / BN, M / BM);          // 32 x 16 workgroups
    wqlinear_wmma_f16<<<grid, dim3(256), 0, stream>>>(x, pw, scale, pzp, bias, out);
    (void)n_in; (void)out_size; (void)d_ws; (void)ws_size;
}
